// TL_classifier_66778151518220
// MI455X (gfx1250) — matrix operand order repaired and checked by reading, not compiled, not run on hardware
//
#include <hip/hip_runtime.h>
#include <math.h>

// ---------------------------------------------------------------------------
// SpaRx-style graph transformer forward for MI455X (gfx1250, wave32, WMMA).
// Heavy GEMMs in f16 via v_wmma_f32_16x16x32_f16 (f32 accumulate).
// ---------------------------------------------------------------------------

#define IN_DIM 15962

typedef _Float16 v16h __attribute__((ext_vector_type(16)));
typedef float    v8f  __attribute__((ext_vector_type(8)));

// ---------------------------------------------------------------------------
// GEMM: C[M][ldc] = A[M][Kp] (f16,row-major) * Bt[Npad][Kp]^T (f16) + bias
// Block: 256 threads (8 wave32), tile 128 x (16*BNT). Each wave owns 16 rows
// and all BNT column tiles. LDS rows padded to 40 f16 (80B) -> 16 fragment
// rows map to distinct banks for the 16B fragment loads.
//
// Pipeline:
//  * ping-pong LDS double buffer, ONE barrier per K-step;
//  * global loads for tile k+1 held in registers during the WMMA chain;
//  * ALL B fragments staged before the WMMA chain (partial s_wait_dscnt);
//  * amdgpu_waves_per_eu(2) raises the VGPR budget (~256/wave) so the staged
//    tiles and fragments stay in registers instead of spilling to scratch
//    (the default occupancy heuristic capped at ~112 VGPRs and spilled).
// ---------------------------------------------------------------------------
template <int BNT>
__global__ void __launch_bounds__(256)
__attribute__((amdgpu_waves_per_eu(2)))
gemm_wmma_kernel(const _Float16* __restrict__ A,
                 const _Float16* __restrict__ Bt,
                 const float* __restrict__ bias,
                 float* __restrict__ C,
                 int M, int N, int Kp, int ldc)
{
    constexpr int BN  = 16 * BNT;
    constexpr int LDT = 40;                  // f16 elems per LDS row (32 + 8 pad)
    constexpr int NB4 = BN * 4;              // 16B chunks in the B tile
    constexpr int CB  = (NB4 + 255) / 256;   // staged B chunks per thread (1 or 2)
    __shared__ __align__(16) _Float16 sA[2][128 * LDT];
    __shared__ __align__(16) _Float16 sB[2][BN  * LDT];

    const int tid  = threadIdx.x;
    const int wave = tid >> 5;
    const int lane = tid & 31;
    const int m0   = blockIdx.x * 128;
    const int n0   = blockIdx.y * BN;

    v8f acc[BNT];
#pragma unroll
    for (int j = 0; j < BNT; ++j)
#pragma unroll
        for (int i = 0; i < 8; ++i) acc[j][i] = 0.f;

    const int r16 = lane & 15;
    const int hi  = lane >> 4;      // 0 or 1
    const int kbA = hi * 8;         // A frag: K {0-7,16-23} / {8-15,24-31}
    const int kbB = hi * 8;         // B frag: K {0-7,16-23} / {8-15,24-31}

    // ---- global -> register tile staging ----
    uint4 ra[2];
    uint4 rb[CB];

    auto loadA = [&](int kk) {
#pragma unroll
        for (int it = 0; it < 2; ++it) {
            int idx = tid + it * 256;         // 0..511
            int r = idx >> 2, c = idx & 3;
            int gm = m0 + r;
            uint4 val = make_uint4(0u, 0u, 0u, 0u);
            if (gm < M) {
                const _Float16* gp = A + (size_t)gm * Kp + kk + c * 8;
                val = *(const uint4*)gp;
                __builtin_prefetch(gp + 32, 0, 1);  // speculative global_prefetch_b8
            }
            ra[it] = val;
        }
    };
    auto loadB = [&](int kk) {
#pragma unroll
        for (int it = 0; it < CB; ++it) {
            int idx = tid + it * 256;
            if ((NB4 % 256 == 0) || idx < NB4) {
                int r = idx >> 2, c = idx & 3;
                rb[it] = *(const uint4*)(Bt + (size_t)(n0 + r) * Kp + kk + c * 8);
            }
        }
    };
    auto stageToLds = [&](int p) {
#pragma unroll
        for (int it = 0; it < 2; ++it) {
            int idx = tid + it * 256;
            int r = idx >> 2, c = idx & 3;
            *(uint4*)(&sA[p][r * LDT + c * 8]) = ra[it];
        }
#pragma unroll
        for (int it = 0; it < CB; ++it) {
            int idx = tid + it * 256;
            if ((NB4 % 256 == 0) || idx < NB4) {
                int r = idx >> 2, c = idx & 3;
                *(uint4*)(&sB[p][r * LDT + c * 8]) = rb[it];
            }
        }
    };

    union Frag { v16h h; uint4 u[2]; };

    loadA(0);
    loadB(0);

    int p = 0;
    for (int kk = 0; kk < Kp; kk += 32, p ^= 1) {
        stageToLds(p);                     // waits on this tile's global loads
        __syncthreads();                   // single barrier per K-step

        // global loads for the NEXT tile fly during the WMMA chain
        if (kk + 32 < Kp) {
            loadA(kk + 32);
            loadB(kk + 32);
        }

        // issue ALL fragment loads first, then the WMMA chain
        Frag af;
        {
            const _Float16* base = &sA[p][(wave * 16 + r16) * LDT + kbA];
            af.u[0] = *(const uint4*)(base);
            af.u[1] = *(const uint4*)(base + 16);
        }
        Frag bf[BNT];
#pragma unroll
        for (int j = 0; j < BNT; ++j) {
            const _Float16* base = &sB[p][(j * 16 + r16) * LDT + kbB];
            bf[j].u[0] = *(const uint4*)(base);
            bf[j].u[1] = *(const uint4*)(base + 16);
        }
#pragma unroll
        for (int j = 0; j < BNT; ++j)
            acc[j] = __builtin_amdgcn_wmma_f32_16x16x32_f16(
                false, af.h, false, bf[j].h, (short)0, acc[j], false, false);
        // no trailing barrier: next iteration stages into the other buffer
    }

    // ---- store with bias; C/D layout: vgpr i -> row i + 8*(lane>>4), col lane&15
    const int col = lane & 15;
    const int ro  = (lane >> 4) * 8;
#pragma unroll
    for (int j = 0; j < BNT; ++j) {
        int n = n0 + j * 16 + col;
        if (n >= N) continue;
        float bv = bias ? bias[n] : 0.f;
#pragma unroll
        for (int i = 0; i < 8; ++i) {
            int m = m0 + wave * 16 + ro + i;
            if (m < M) C[(size_t)m * ldc + n] = acc[j][i] + bv;
        }
    }
}

// ---------------------------------------------------------------------------
// Attention helpers (segment softmax over incoming edges per target node)
// ---------------------------------------------------------------------------
__device__ __forceinline__ unsigned fkey(float f) {
    unsigned b = __float_as_uint(f);
    return (b >> 31) ? ~b : (b | 0x80000000u);    // order-preserving encode
}
__device__ __forceinline__ float unfkey(unsigned k) {
    return (k >> 31) ? __uint_as_float(k & 0x7fffffffu) : __uint_as_float(~k);
}

__global__ void attn_init_kernel(unsigned* __restrict__ mkey,
                                 float* __restrict__ denom, int M)
{
    int t = blockIdx.x * blockDim.x + threadIdx.x;
    if (t < M) { mkey[t] = 0u; denom[t] = 0.f; }
}

// wave per edge: logits[e] = dot(q[dst], k[src]) * scale ; segment max
__global__ void edge_logits_kernel(const float* __restrict__ q,
                                   const float* __restrict__ k,
                                   const int* __restrict__ sI,
                                   const int* __restrict__ dI,
                                   float* __restrict__ logits,
                                   unsigned* __restrict__ mkey,
                                   int E, int Cdim, int ld, float scale)
{
    int e    = (blockIdx.x * blockDim.x + threadIdx.x) >> 5;
    int lane = threadIdx.x & 31;
    if (e >= E) return;
    int s = sI[e], d = dI[e];
    float acc = 0.f;
    for (int c = lane; c < Cdim; c += 32)
        acc += q[(size_t)d * ld + c] * k[(size_t)s * ld + c];
#pragma unroll
    for (int off = 16; off; off >>= 1)
        acc += __shfl_xor(acc, off, 32);
    if (lane == 0) {
        float l = acc * scale;
        logits[e] = l;
        atomicMax(&mkey[d], fkey(l));
    }
}

__global__ void edge_exp_kernel(const float* __restrict__ logits,
                                const int* __restrict__ dI,
                                const unsigned* __restrict__ mkey,
                                float* __restrict__ exv,
                                float* __restrict__ denom, int E)
{
    int e = blockIdx.x * blockDim.x + threadIdx.x;
    if (e >= E) return;
    int d = dI[e];
    float x = __expf(logits[e] - unfkey(mkey[d]));
    exv[e] = x;
    atomicAdd(&denom[d], x);
}

// wave per edge: out[dst] += (ex/denom) * v[src]   (out already holds skip)
__global__ void edge_agg_kernel(const float* __restrict__ exv,
                                const float* __restrict__ denom,
                                const float* __restrict__ v,
                                const int* __restrict__ sI,
                                const int* __restrict__ dI,
                                float* __restrict__ out,
                                int E, int Cdim, int ld)
{
    int e    = (blockIdx.x * blockDim.x + threadIdx.x) >> 5;
    int lane = threadIdx.x & 31;
    if (e >= E) return;
    int s = sI[e], d = dI[e];
    float a = exv[e] / (denom[d] + 1e-16f);
    for (int c = lane; c < Cdim; c += 32)
        atomicAdd(&out[(size_t)d * ld + c], a * v[(size_t)s * ld + c]);
}

// ---------------------------------------------------------------------------
// Elementwise: act (0=none,1=relu,2=elu), optional f32 store (packed, ld32)
// and optional f16 store with zero-padded width Wd (next layer's Kpad)
// ---------------------------------------------------------------------------
__global__ void act_convert_kernel(const float* __restrict__ in, int ldin,
                                   int M, int N, int Wd,
                                   float* __restrict__ f32out, int ld32,
                                   _Float16* __restrict__ f16out, int mode)
{
    long t = (long)blockIdx.x * blockDim.x + threadIdx.x;
    if (t >= (long)M * Wd) return;
    int m = (int)(t / Wd), c = (int)(t % Wd);
    float val = 0.f;
    if (c < N) {
        float x = in[(size_t)m * ldin + c];
        if (mode == 1)      val = x > 0.f ? x : 0.f;
        else if (mode == 2) val = x > 0.f ? x : (__expf(x) - 1.f);
        else                val = x;
    }
    if (f16out) f16out[(size_t)m * Wd + c] = (_Float16)val;
    if (f32out && c < N) f32out[(size_t)m * ld32 + c] = val;
}

// f32 [M][K] -> f16 [M][Kp], zero K-pad
__global__ void x_convert_kernel(const float* __restrict__ x,
                                 _Float16* __restrict__ xh,
                                 int M, int K, int Kp)
{
    long t = (long)blockIdx.x * blockDim.x + threadIdx.x;
    if (t >= (long)M * Kp) return;
    int m = (int)(t / Kp), c = (int)(t % Kp);
    xh[(size_t)m * Kp + c] = (c < K) ? (_Float16)x[(size_t)m * K + c] : (_Float16)0.f;
}

// W f32 [K][N] (row-major) -> Wt f16 [Np][Kp] (transposed, zero-padded)
__global__ void w_convert_kernel(const float* __restrict__ W,
                                 _Float16* __restrict__ Wt,
                                 int K, int N, int Kp, int Np)
{
    long t = (long)blockIdx.x * blockDim.x + threadIdx.x;
    if (t >= (long)Np * Kp) return;
    int n = (int)(t / Kp), c = (int)(t % Kp);
    float v = (n < N && c < K) ? W[(size_t)c * N + n] : 0.f;
    Wt[(size_t)n * Kp + c] = (_Float16)v;
}

// fsh[m][c] = softmax(pred[m])[cls] * emb[m][c], as f16 (C == Kpad == 64)
__global__ void fs_convert_kernel(const float* __restrict__ pred,
                                  const float* __restrict__ emb,
                                  _Float16* __restrict__ fsh,
                                  int M, int Cdim, int cls)
{
    int t = blockIdx.x * blockDim.x + threadIdx.x;
    if (t >= M * Cdim) return;
    int m = t / Cdim, c = t % Cdim;
    float a0 = pred[(size_t)m * 2], a1 = pred[(size_t)m * 2 + 1];
    float mx = fmaxf(a0, a1);
    float e0 = __expf(a0 - mx), e1 = __expf(a1 - mx);
    float p  = (cls == 0 ? e0 : e1) / (e0 + e1);
    fsh[(size_t)m * Cdim + c] = (_Float16)(p * emb[(size_t)m * Cdim + c]);
}

// ---------------------------------------------------------------------------
extern "C" void kernel_launch(void* const* d_in, const int* in_sizes, int n_in,
                              void* d_out, int out_size, void* d_ws, size_t ws_size,
                              hipStream_t stream)
{
    (void)n_in; (void)out_size; (void)ws_size;
    const int M = in_sizes[0] / IN_DIM;               // nodes per graph (4000)

    // Input map (insertion-order pytree flatten of setup_inputs()):
    //   0:src_x 1:tar_x
    //   params: share(2 convs), src_cls(3), dom_cls(3), dcis[0](3), dcis[1](3)
    //     each conv: Wq,bq,Wk,bk,Wv,bv,Ws,bs  (8 leaves)
    //   then src_edge, tar_edge, alpha
    const int IDX_EDGE_SRC = 2 + 8 * 14;              // 114
    const int IDX_EDGE_TAR = IDX_EDGE_SRC + 1;
    const int E = in_sizes[IDX_EDGE_SRC] / 2;

    struct Spec { int idx, din, dout; };
    const Spec specs[14] = {
        {2,  IN_DIM, 512}, {10, 512, 64},             // share
        {18, 64, 32}, {26, 32, 16}, {34, 16, 2},      // src_cls
        {42, 64, 32}, {50, 32, 16}, {58, 16, 2},      // dom_cls
        {66, 64, 32}, {74, 32, 16}, {82, 16, 2},      // dcis[0]
        {90, 64, 32}, {98, 32, 16}, {106,16, 2},      // dcis[1]
    };

    char* wsb = (char*)d_ws;
    size_t off = 0;
    auto alloc = [&](size_t bytes) -> void* {
        off = (off + 255) & ~(size_t)255;
        void* p = wsb + off;
        off += bytes;
        return p;
    };

    struct LayerW {
        const _Float16 *q, *k, *v, *s;
        const float *bq, *bk, *bv, *bs;
        int din, dout, Kp, Np;
    };
    LayerW L[14];

    // ---- one-shot weight transpose/convert to f16 [Np][Kp] ----
    for (int li = 0; li < 14; ++li) {
        const Spec& sp = specs[li];
        int Kp = (sp.din + 31) & ~31;
        int Np = (sp.dout + 15) & ~15;
        size_t msz = (size_t)Np * Kp;
        _Float16* wb = (_Float16*)alloc(4 * msz * sizeof(_Float16));
        L[li].q = wb;           L[li].k = wb + msz;
        L[li].v = wb + 2*msz;   L[li].s = wb + 3*msz;
        L[li].bq = (const float*)d_in[sp.idx + 1];
        L[li].bk = (const float*)d_in[sp.idx + 3];
        L[li].bv = (const float*)d_in[sp.idx + 5];
        L[li].bs = (const float*)d_in[sp.idx + 7];
        L[li].din = sp.din; L[li].dout = sp.dout; L[li].Kp = Kp; L[li].Np = Np;
        const int wi[4] = { sp.idx + 0, sp.idx + 2, sp.idx + 4, sp.idx + 6 };
        _Float16* dsts[4] = { wb, wb + msz, wb + 2*msz, wb + 3*msz };
        long tot = (long)Np * Kp;
        for (int w = 0; w < 4; ++w)
            w_convert_kernel<<<(unsigned)((tot + 255) / 256), 256, 0, stream>>>(
                (const float*)d_in[wi[w]], dsts[w], sp.din, sp.dout, Kp, Np);
    }

    // ---- scratch buffers ----
    const int KP1 = (IN_DIM + 31) & ~31;              // 15968
    _Float16* Xh    = (_Float16*)alloc((size_t)M * KP1 * 2);
    _Float16* h1h   = (_Float16*)alloc((size_t)M * 512 * 2);
    _Float16* embhS = (_Float16*)alloc((size_t)M * 64 * 2);
    _Float16* embhT = (_Float16*)alloc((size_t)M * 64 * 2);
    _Float16* fsh   = (_Float16*)alloc((size_t)M * 64 * 2);
    _Float16* s32h  = (_Float16*)alloc((size_t)M * 32 * 2);
    _Float16* s16h  = (_Float16*)alloc((size_t)M * 32 * 2);
    float* bufQ  = (float*)alloc((size_t)M * 512 * 4);
    float* bufK  = (float*)alloc((size_t)M * 512 * 4);
    float* bufV  = (float*)alloc((size_t)M * 512 * 4);
    float* bufO  = (float*)alloc((size_t)M * 512 * 4);
    unsigned* mkey = (unsigned*)alloc((size_t)M * 4);
    float* denom   = (float*)alloc((size_t)M * 4);
    float* logits  = (float*)alloc((size_t)E * 4);
    float* exv     = (float*)alloc((size_t)E * 4);
    float* predT   = (float*)alloc((size_t)M * 2 * 4);   // tar_pred (not an output)

    auto gemm = [&](const _Float16* A, const _Float16* Bt, const float* bias,
                    float* Cp, int N, int Kp, int Np) {
        unsigned gx = (unsigned)((M + 127) / 128);
        if (Np % 128 == 0)
            gemm_wmma_kernel<8><<<dim3(gx, Np / 128), 256, 0, stream>>>(A, Bt, bias, Cp, M, N, Kp, Np);
        else if (Np == 64)
            gemm_wmma_kernel<4><<<dim3(gx, 1), 256, 0, stream>>>(A, Bt, bias, Cp, M, N, Kp, Np);
        else if (Np == 32)
            gemm_wmma_kernel<2><<<dim3(gx, 1), 256, 0, stream>>>(A, Bt, bias, Cp, M, N, Kp, Np);
        else
            gemm_wmma_kernel<1><<<dim3(gx, 1), 256, 0, stream>>>(A, Bt, bias, Cp, M, N, Kp, Np);
    };

    auto run_conv = [&](const LayerW& lw, const _Float16* xh,
                        const int* sI, const int* dI,
                        int actmode, float* f32out, int ld32,
                        _Float16* f16out, int outKp) {
        gemm(xh, lw.q, lw.bq, bufQ, lw.dout, lw.Kp, lw.Np);
        gemm(xh, lw.k, lw.bk, bufK, lw.dout, lw.Kp, lw.Np);
        gemm(xh, lw.v, lw.bv, bufV, lw.dout, lw.Kp, lw.Np);
        gemm(xh, lw.s, lw.bs, bufO, lw.dout, lw.Kp, lw.Np);   // skip term
        attn_init_kernel<<<(M + 255) / 256, 256, 0, stream>>>(mkey, denom, M);
        float scale = 1.f / sqrtf((float)lw.dout);
        edge_logits_kernel<<<(E * 32 + 255) / 256, 256, 0, stream>>>(
            bufQ, bufK, sI, dI, logits, mkey, E, lw.dout, lw.Np, scale);
        edge_exp_kernel<<<(E + 255) / 256, 256, 0, stream>>>(logits, dI, mkey, exv, denom, E);
        edge_agg_kernel<<<(E * 32 + 255) / 256, 256, 0, stream>>>(
            exv, denom, bufV, sI, dI, bufO, E, lw.dout, lw.Np);
        int Wd = f16out ? outKp : lw.dout;
        long tot = (long)M * Wd;
        act_convert_kernel<<<(unsigned)((tot + 255) / 256), 256, 0, stream>>>(
            bufO, lw.Np, M, lw.dout, Wd, f32out, ld32, f16out, actmode);
    };

    // ---- d_out layout (return-order concat) ----
    float* out = (float*)d_out;
    float* o_src_pred = out;
    float* o_src_dom  = out + (size_t)2 * M;
    float* o_tar_dom  = out + (size_t)4 * M;
    float* o_s0       = out + (size_t)6 * M;
    float* o_s1       = out + (size_t)8 * M;
    float* o_t0       = out + (size_t)10 * M;
    float* o_t1       = out + (size_t)12 * M;
    float* o_src_emb  = out + (size_t)14 * M;
    float* o_tar_emb  = out + (size_t)14 * M + (size_t)64 * M;

    auto run_graph = [&](const float* X, const int* edge, _Float16* embh,
                         float* embOut, float* predOut, float* domOut,
                         float* dciOut0, float* dciOut1) {
        const int* sI = edge;
        const int* dI = edge + E;
        long tot = (long)M * KP1;
        x_convert_kernel<<<(unsigned)((tot + 255) / 256), 256, 0, stream>>>(X, Xh, M, IN_DIM, KP1);
        // shared feature extractor (ELU)
        run_conv(L[0], Xh,  sI, dI, 2, nullptr, 0, h1h, 512);
        run_conv(L[1], h1h, sI, dI, 2, embOut, 64, embh, 64);
        // class predictor (ReLU incl. last layer)
        run_conv(L[2], embh, sI, dI, 1, nullptr, 0, s32h, 32);
        run_conv(L[3], s32h, sI, dI, 1, nullptr, 0, s16h, 32);
        run_conv(L[4], s16h, sI, dI, 1, predOut, 2, nullptr, 0);
        // domain classifier (grad-reversal is identity in forward)
        run_conv(L[5], embh, sI, dI, 1, nullptr, 0, s32h, 32);
        run_conv(L[6], s32h, sI, dI, 1, nullptr, 0, s16h, 32);
        run_conv(L[7], s16h, sI, dI, 1, domOut, 2, nullptr, 0);
        // per-class conditional domain classifiers
        float* dciOut[2] = { dciOut0, dciOut1 };
        for (int ci = 0; ci < 2; ++ci) {
            fs_convert_kernel<<<(unsigned)((M * 64 + 255) / 256), 256, 0, stream>>>(
                predOut, embOut, fsh, M, 64, ci);
            int b = 8 + 3 * ci;
            run_conv(L[b + 0], fsh,  sI, dI, 1, nullptr, 0, s32h, 32);
            run_conv(L[b + 1], s32h, sI, dI, 1, nullptr, 0, s16h, 32);
            run_conv(L[b + 2], s16h, sI, dI, 1, dciOut[ci], 2, nullptr, 0);
        }
    };

    run_graph((const float*)d_in[0], (const int*)d_in[IDX_EDGE_SRC],
              embhS, o_src_emb, o_src_pred, o_src_dom, o_s0, o_s1);
    run_graph((const float*)d_in[1], (const int*)d_in[IDX_EDGE_TAR],
              embhT, o_tar_emb, predT,      o_tar_dom, o_t0, o_t1);
}
